// MultiHeadAttention_34754875359731
// MI455X (gfx1250) — matrix operand order repaired and checked by reading, not compiled, not run on hardware
//
#include <hip/hip_runtime.h>
#include <hip/hip_bf16.h>

// ---------------------------------------------------------------------------
// MHA forward for MI455X (gfx1250, wave32, WMMA).
//   1) gemm_proj x3 : Q/K/V projections (fp32 in, bf16 head-layout out)
//        Q,K -> (b,h,s,64)    V -> (b,h,64,s)  (transposed for ctx GEMM)
//   2) attn_softmax : per (b,h,16q) logits via v_wmma_f32_16x16x32_bf16,
//        K fragments straight from global, masked softmax, attn fp32 -> d_out
//   3) ctx_gemm     : ctx = attn @ V (bf16 WMMA; V tile via async LDS copy)
//   4) gemm_proj    : out = ctx @ wo + bo (fp32 -> d_out)
// ---------------------------------------------------------------------------

#define SEQ   1024
#define NH    16
#define DH    64
#define BATCH 8
#define DMODEL 1024

typedef __attribute__((ext_vector_type(16))) __bf16 bf16x16;
typedef __attribute__((ext_vector_type(8)))  __bf16 bf16x8;
typedef __attribute__((ext_vector_type(4)))  __bf16 bf16x4;
typedef __attribute__((ext_vector_type(8)))  float  f32x8;
typedef __attribute__((ext_vector_type(4)))  float  f32x4;
typedef __attribute__((ext_vector_type(4)))  int    v4i;

union Frag { bf16x16 v; bf16x8 h[2]; };

static __device__ __forceinline__ f32x8 wmma_bf16(const Frag& a, const Frag& b, f32x8 c) {
    // D = A(16x32 bf16) * B(32x16 bf16) + C(16x16 f32)
    return __builtin_amdgcn_wmma_f32_16x16x32_bf16(
        /*neg_a=*/false, a.v, /*neg_b=*/false, b.v,
        /*c_mod=*/(short)0, c, /*reuse_a=*/false, /*reuse_b=*/false);
}

static __device__ __forceinline__ bf16x4 cvt4(f32x4 f) {
    bf16x4 r;
    r[0] = (__bf16)f[0]; r[1] = (__bf16)f[1];
    r[2] = (__bf16)f[2]; r[3] = (__bf16)f[3];
    return r;
}

// ---- gfx1250 async global->LDS copy (ASYNCcnt path), guarded fallback ------
#if defined(__HIP_DEVICE_COMPILE__) && \
    __has_builtin(__builtin_amdgcn_global_load_async_to_lds_b128)
#define HAVE_ASYNC_LDS 1
#else
#define HAVE_ASYNC_LDS 0
#endif

#define AS1 __attribute__((address_space(1)))
#define AS3 __attribute__((address_space(3)))

static __device__ __forceinline__ void copy16_g2l(const __bf16* g, __bf16* l) {
#if HAVE_ASYNC_LDS
    __builtin_amdgcn_global_load_async_to_lds_b128(
        (AS1 v4i*)(uintptr_t)g,
        (AS3 v4i*)(unsigned)(uintptr_t)l, 0, 0);
#else
    *(bf16x8*)l = *(const bf16x8*)g;
#endif
}

static __device__ __forceinline__ void wait_async_copies() {
#if HAVE_ASYNC_LDS
#if __has_builtin(__builtin_amdgcn_s_wait_asynccnt)
    __builtin_amdgcn_s_wait_asynccnt(0);
#else
    asm volatile("s_wait_asynccnt 0" ::: "memory");
#endif
#endif
}

// ---------------------------------------------------------------------------
// Generic tiled GEMM: C[MxN] = A[MxK] (fp32) * B[KxN] (fp32) + bias
// Block = 128 threads (4 waves), tile 64x64, BK=32. Each wave: 32x32 (2x2 WMMA).
// mode 0: fp32 row-major -> outf
// mode 1: bf16 head-split (b,h,s,d)            -> outb
// mode 2: bf16 head-split transposed (b,h,d,s) -> outb
// ---------------------------------------------------------------------------
__global__ __launch_bounds__(128)
void gemm_proj(const float* __restrict__ A, const float* __restrict__ Bm,
               const float* __restrict__ bias, float* __restrict__ outf,
               __bf16* __restrict__ outb, int M, int N, int K, int mode)
{
    __shared__ __bf16 As[64 * 32];   // [row][k] row-major
    __shared__ __bf16 Bs[64 * 32];   // [n][k]  (B transposed -> k contiguous)

    const int t    = threadIdx.x;
    const int w    = t >> 5, lane = t & 31;
    const int half = lane >> 4, l16 = lane & 15;
    const int rowblk = blockIdx.x * 64;
    const int colblk = blockIdx.y * 64;
    const int wr = (w >> 1) * 32, wc = (w & 1) * 32;

    // 32-bit offsets within matrices (A <= 8192x1024, B <= 1024x1024 floats)
    const float* Abase = A + (size_t)rowblk * K;
    const float* Bbase = Bm + colblk;

    f32x8 acc[2][2] = {};

    for (int kb = 0; kb < K; kb += 32) {
        // A tile 64x32: one b128 load + one b64 LDS store per thread per iter
        #pragma unroll
        for (int it = 0; it < 4; ++it) {
            int idx = t + it * 128;
            int r = idx >> 3, c4 = (idx & 7) * 4;
            f32x4 f = *(const f32x4*)&Abase[r * K + kb + c4];
            *(bf16x4*)&As[r * 32 + c4] = cvt4(f);
        }
        // B tile 32x64 transposed: 4 coalesced dword loads, one b64 LDS store
        #pragma unroll
        for (int it = 0; it < 4; ++it) {
            int idx = t + it * 128;
            int n = idx & 63, k4 = (idx >> 6) * 4;
            f32x4 f;
            #pragma unroll
            for (int j = 0; j < 4; ++j)
                f[j] = Bbase[(kb + k4 + j) * N + n];
            *(bf16x4*)&Bs[n * 32 + k4] = cvt4(f);
        }
        __syncthreads();

        Frag af[2], bf_[2];
        #pragma unroll
        for (int m = 0; m < 2; ++m) {
            int r = wr + m * 16 + l16;
            af[m].h[0] = *(const bf16x8*)&As[r * 32 + half * 8];
            af[m].h[1] = *(const bf16x8*)&As[r * 32 + 16 + half * 8];
        }
        #pragma unroll
        for (int n = 0; n < 2; ++n) {
            int c = wc + n * 16 + l16;
            bf_[n].h[0] = *(const bf16x8*)&Bs[c * 32 + half * 8];
            bf_[n].h[1] = *(const bf16x8*)&Bs[c * 32 + 16 + half * 8];
        }
        #pragma unroll
        for (int m = 0; m < 2; ++m)
            #pragma unroll
            for (int n = 0; n < 2; ++n)
                acc[m][n] = wmma_bf16(af[m], bf_[n], acc[m][n]);
        __syncthreads();
    }

    #pragma unroll
    for (int m = 0; m < 2; ++m)
        #pragma unroll
        for (int n = 0; n < 2; ++n) {
            int baseR = rowblk + wr + m * 16 + half * 8;
            int c     = colblk + wc + n * 16 + l16;
            float bv  = bias ? bias[c] : 0.0f;
            #pragma unroll
            for (int i = 0; i < 8; ++i) {
                int r = baseR + i;
                float vv = acc[m][n][i] + bv;
                if (mode == 0) {
                    outf[(size_t)r * N + c] = vv;
                } else {
                    int b = r / SEQ, s = r % SEQ;
                    int hh = c / DH, d = c % DH;
                    if (mode == 1)
                        outb[(((size_t)(b * NH + hh)) * SEQ + s) * DH + d] = (__bf16)vv;
                    else
                        outb[(((size_t)(b * NH + hh)) * DH + d) * SEQ + s] = (__bf16)vv;
                }
            }
        }
}

// ---------------------------------------------------------------------------
// Attention logits + softmax. Block = 256 threads (8 waves) handles one
// (b, h, 16-query tile). Wave w covers keys [w*128, w*128+128) = 8 WMMA tiles.
// Q/K fragments load straight from global (bf16 head-layout == frag layout).
// ---------------------------------------------------------------------------
__global__ __launch_bounds__(256)
void attn_softmax(const __bf16* __restrict__ qh, const __bf16* __restrict__ kh,
                  const float* __restrict__ mask, float* __restrict__ attn)
{
    __shared__ float red[2][8][16];

    const int blk = blockIdx.x;
    const int qt = blk & 63;            // SEQ/16
    const int hh = (blk >> 6) & (NH - 1);
    const int b  = blk >> 10;
    const int t = threadIdx.x, w = t >> 5, lane = t & 31;
    const int half = lane >> 4, l16 = lane & 15;

    const size_t bh = (size_t)b * NH + hh;
    const __bf16* qp = qh + (bh * SEQ + (size_t)qt * 16) * DH;
    const __bf16* kp = kh + bh * SEQ * DH;

    f32x8 acc[8] = {};
    #pragma unroll
    for (int ds = 0; ds < DH; ds += 32) {
        Frag a;
        a.h[0] = *(const bf16x8*)&qp[l16 * DH + ds + half * 8];
        a.h[1] = *(const bf16x8*)&qp[l16 * DH + ds + 16 + half * 8];
        #pragma unroll
        for (int kt = 0; kt < 8; ++kt) {
            int key = w * 128 + kt * 16 + l16;
            Frag bb;
            bb.h[0] = *(const bf16x8*)&kp[key * DH + ds + half * 8];
            bb.h[1] = *(const bf16x8*)&kp[key * DH + ds + 16 + half * 8];
            acc[kt] = wmma_bf16(a, bb, acc[kt]);
        }
    }

    // scale (1/sqrt(64)) + additive mask
    const float scale = 0.125f;
    #pragma unroll
    for (int kt = 0; kt < 8; ++kt) {
        int col = w * 128 + kt * 16 + l16;
        float mv = mask[(size_t)b * SEQ + col] * -1e9f;
        #pragma unroll
        for (int i = 0; i < 8; ++i) acc[kt][i] = acc[kt][i] * scale + mv;
    }

    // row max: per-lane over 8 tiles, across 16 lanes (wave32 shfl), across waves
    float pv[8];
    #pragma unroll
    for (int i = 0; i < 8; ++i) {
        float m = acc[0][i];
        #pragma unroll
        for (int kt = 1; kt < 8; ++kt) m = fmaxf(m, acc[kt][i]);
        #pragma unroll
        for (int off = 1; off < 16; off <<= 1) m = fmaxf(m, __shfl_xor(m, off, 32));
        pv[i] = m;
    }
    if (l16 == 0) {
        #pragma unroll
        for (int i = 0; i < 8; ++i) red[0][w][half * 8 + i] = pv[i];
    }
    __syncthreads();
    float rstat[8];
    #pragma unroll
    for (int i = 0; i < 8; ++i) {
        float m = red[0][0][half * 8 + i];
        #pragma unroll
        for (int ww = 1; ww < 8; ++ww) m = fmaxf(m, red[0][ww][half * 8 + i]);
        rstat[i] = m;
    }

    // exp + row sum
    float ps[8];
    #pragma unroll
    for (int i = 0; i < 8; ++i) ps[i] = 0.0f;
    #pragma unroll
    for (int kt = 0; kt < 8; ++kt)
        #pragma unroll
        for (int i = 0; i < 8; ++i) {
            float e = __expf(acc[kt][i] - rstat[i]);
            acc[kt][i] = e;
            ps[i] += e;
        }
    #pragma unroll
    for (int i = 0; i < 8; ++i)
        #pragma unroll
        for (int off = 1; off < 16; off <<= 1) ps[i] += __shfl_xor(ps[i], off, 32);
    if (l16 == 0) {
        #pragma unroll
        for (int i = 0; i < 8; ++i) red[1][w][half * 8 + i] = ps[i];
    }
    __syncthreads();
    #pragma unroll
    for (int i = 0; i < 8; ++i) {
        float s = 0.0f;
        #pragma unroll
        for (int ww = 0; ww < 8; ++ww) s += red[1][ww][half * 8 + i];
        rstat[i] = 1.0f / s;   // reuse as inverse sum
    }

    // write normalized probabilities
    float* ap = attn + (bh * SEQ + (size_t)qt * 16) * SEQ;
    #pragma unroll
    for (int kt = 0; kt < 8; ++kt) {
        int col = w * 128 + kt * 16 + l16;
        #pragma unroll
        for (int i = 0; i < 8; ++i) {
            int r = half * 8 + i;
            ap[(size_t)r * SEQ + col] = acc[kt][i] * rstat[i];
        }
    }
}

// ---------------------------------------------------------------------------
// ctx = attn @ V   per (b,h): M=1024 (q), N=64 (d), K=1024 (keys)
// attn fp32 -> bf16 in LDS; V tile copied with GLOBAL_LOAD_ASYNC_TO_LDS_B128
// (already bf16, (d,s) layout == transposed-B layout).
// Output ctx fp32 at (b*SEQ+q, h*64+d) in an 8192x1024 row-major buffer.
// ---------------------------------------------------------------------------
__global__ __launch_bounds__(128)
void ctx_gemm(const float* __restrict__ attn, const __bf16* __restrict__ vt,
              float* __restrict__ ctx)
{
    __shared__ __bf16 As[64 * 32];
    __shared__ __bf16 Bs[64 * 32];

    const int blk = blockIdx.x;
    const int mt = blk & 15;            // SEQ/64
    const int hh = (blk >> 4) & (NH - 1);
    const int b  = blk >> 8;
    const int t = threadIdx.x, w = t >> 5, lane = t & 31;
    const int half = lane >> 4, l16 = lane & 15;
    const int wr = (w >> 1) * 32, wc = (w & 1) * 32;

    const size_t bh = (size_t)b * NH + hh;
    const float*  ap = attn + bh * SEQ * SEQ + (size_t)mt * 64 * SEQ;
    const __bf16* vp = vt + bh * DH * SEQ;

    f32x8 acc[2][2] = {};
    for (int kb = 0; kb < SEQ; kb += 32) {
        // attn tile 64x32 fp32 -> bf16 (b128 load + b64 LDS store)
        #pragma unroll
        for (int it = 0; it < 4; ++it) {
            int idx = t + it * 128;
            int r = idx >> 3, c4 = (idx & 7) * 4;
            f32x4 f = *(const f32x4*)&ap[r * SEQ + kb + c4];
            *(bf16x4*)&As[r * 32 + c4] = cvt4(f);
        }
        // V tile 64(d) x 32(k): pure 16B copies -> async LDS path
        #pragma unroll
        for (int it = 0; it < 2; ++it) {
            int idx = t + it * 128;
            int n = idx >> 2, k8 = (idx & 3) * 8;
            copy16_g2l(&vp[n * SEQ + kb + k8], &Bs[n * 32 + k8]);
        }
        wait_async_copies();
        __syncthreads();

        Frag af[2], bf_[2];
        #pragma unroll
        for (int m = 0; m < 2; ++m) {
            int r = wr + m * 16 + l16;
            af[m].h[0] = *(const bf16x8*)&As[r * 32 + half * 8];
            af[m].h[1] = *(const bf16x8*)&As[r * 32 + 16 + half * 8];
        }
        #pragma unroll
        for (int n = 0; n < 2; ++n) {
            int c = wc + n * 16 + l16;
            bf_[n].h[0] = *(const bf16x8*)&Bs[c * 32 + half * 8];
            bf_[n].h[1] = *(const bf16x8*)&Bs[c * 32 + 16 + half * 8];
        }
        #pragma unroll
        for (int m = 0; m < 2; ++m)
            #pragma unroll
            for (int n = 0; n < 2; ++n)
                acc[m][n] = wmma_bf16(af[m], bf_[n], acc[m][n]);
        __syncthreads();
    }

    #pragma unroll
    for (int m = 0; m < 2; ++m)
        #pragma unroll
        for (int n = 0; n < 2; ++n) {
            int baseR = b * SEQ + mt * 64 + wr + m * 16 + half * 8;
            int c     = hh * DH + wc + n * 16 + l16;
            #pragma unroll
            for (int i = 0; i < 8; ++i)
                ctx[(size_t)(baseR + i) * (NH * DH) + c] = acc[m][n][i];
        }
}

// ---------------------------------------------------------------------------
extern "C" void kernel_launch(void* const* d_in, const int* in_sizes, int n_in,
                              void* d_out, int out_size, void* d_ws, size_t ws_size,
                              hipStream_t stream) {
    const float* q    = (const float*)d_in[0];
    const float* k    = (const float*)d_in[1];
    const float* v    = (const float*)d_in[2];
    const float* mask = (const float*)d_in[3];
    const float* wq   = (const float*)d_in[4];
    const float* bq   = (const float*)d_in[5];
    const float* wk   = (const float*)d_in[6];
    const float* bk   = (const float*)d_in[7];
    const float* wv   = (const float*)d_in[8];
    const float* bv   = (const float*)d_in[9];
    const float* wo   = (const float*)d_in[10];
    const float* bo   = (const float*)d_in[11];

    float* out  = (float*)d_out;                               // (B,SQ,1024)
    float* attn = out + (size_t)BATCH * SEQ * DMODEL;          // (B,H,SQ,SK)

    char* ws = (char*)d_ws;
    __bf16* qh  = (__bf16*)ws;                                 // (B,H,S,64) 16 MiB
    __bf16* kh  = (__bf16*)(ws + (size_t)16 * 1024 * 1024);    // (B,H,S,64) 16 MiB
    __bf16* vt  = (__bf16*)(ws + (size_t)32 * 1024 * 1024);    // (B,H,64,S) 16 MiB
    float*  ctx = (float*) (ws + (size_t)48 * 1024 * 1024);    // (8192,1024) 32 MiB

    const int M = BATCH * SEQ;          // 8192
    dim3 gproj(M / 64, DMODEL / 64);    // 128 x 16

    gemm_proj<<<gproj, 128, 0, stream>>>(q, wq, bq, nullptr, qh, M, DMODEL, DMODEL, 1);
    gemm_proj<<<gproj, 128, 0, stream>>>(k, wk, bk, nullptr, kh, M, DMODEL, DMODEL, 1);
    gemm_proj<<<gproj, 128, 0, stream>>>(v, wv, bv, nullptr, vt, M, DMODEL, DMODEL, 2);

    attn_softmax<<<BATCH * NH * (SEQ / 16), 256, 0, stream>>>(qh, kh, mask, attn);

    ctx_gemm<<<BATCH * NH * (SEQ / 64), 128, 0, stream>>>(attn, vt, ctx);

    gemm_proj<<<gproj, 128, 0, stream>>>(ctx, wo, bo, out, nullptr, M, DMODEL, DMODEL, 0);
}
